// Model_51410758533207
// MI455X (gfx1250) — matrix operand order repaired and checked by reading, not compiled, not run on hardware
//
#include <hip/hip_runtime.h>
#include <hip/hip_bf16.h>

#define N_NODES  100000
#define N_EDGES  250000
#define G_GRAPHS 4096
#define DIM      300
#define HID      600
#define DIMP     320     // DIM padded to multiple of 32/64
#define HIDP     640     // HID padded to multiple of 32/64
#define LAYERS   5
#define BN_EPS   1e-5f

typedef __bf16 bf16_t;
typedef __attribute__((ext_vector_type(16))) __bf16 v16bfx;
typedef __attribute__((ext_vector_type(8)))  __bf16 v8bfx;   // 16 bytes
typedef __attribute__((ext_vector_type(8)))  float  v8fx;
typedef __attribute__((ext_vector_type(4)))  int    v4ix;    // 16 bytes

#if __has_builtin(__builtin_amdgcn_global_load_async_to_lds_b128) && \
    __has_builtin(__builtin_amdgcn_s_wait_asynccnt)
#define HAVE_ASYNC_LDS 1
#define AS1(p) ((__attribute__((address_space(1))) v4ix*)(p))
#define AS3(p) ((__attribute__((address_space(3))) v4ix*)(p))
#else
#define HAVE_ASYNC_LDS 0
#endif

static inline int ceildiv_i(int a, int b) { return (a + b - 1) / b; }

// ======================================================================
// WMMA bf16 GEMM:  C[M,N] = A[M,K] * B[K,N] + bias
// K, N, lda, ldb, ldc padded (K%32==0, N%64==0), bias padded -> staging
// has no K/N guards. Double-buffered LDS, bank-conflict padded, peeled
// K-loop, async global->LDS staging when the toolchain provides it.
// mode 0: C = bf16, relu(x + bias);  mode 1: C = f32, x + bias
// ======================================================================
#define BM 128
#define BN 64
#define BK 32
#define AS_LD 40   // BK + 8 (bank-conflict pad)
#define BS_LD 72   // BN + 8

__global__ __launch_bounds__(256)
void gemm_bf16_wmma_kernel(const bf16_t* __restrict__ A, int lda,
                           const bf16_t* __restrict__ B, int ldb,
                           const float* __restrict__ bias,
                           const bf16_t* __restrict__ zsrc,  // 16B of zeros
                           int M, int K, int N,
                           bf16_t* __restrict__ Cb, float* __restrict__ Cf,
                           int ldc, int mode) {
  __shared__ bf16_t As[2][BM][AS_LD];
  __shared__ bf16_t Bs[2][BK][BS_LD];

  const int tid  = threadIdx.x;
  const int wave = tid >> 5;
  const int lane = tid & 31;
  const int m0   = blockIdx.y * BM;
  const int n0   = blockIdx.x * BN;
  const int wm   = m0 + wave * 16;

  v8bfx zero8;
#pragma unroll
  for (int j = 0; j < 8; ++j) zero8[j] = (bf16_t)0.0f;

  v8fx acc[4];
#pragma unroll
  for (int t = 0; t < 4; ++t)
#pragma unroll
    for (int j = 0; j < 8; ++j) acc[t][j] = 0.0f;

  // ---- staging: per-lane 16B global->LDS copies ----
  auto stage = [&](int buf, int k0) {
    // A tile: 128 rows x 32 cols = 512 x 16B, 2 per thread
#pragma unroll
    for (int v = tid; v < (BM * BK) / 8; v += 256) {
      int r = v >> 2;            // row in tile
      int c = (v & 3) << 3;      // col (multiple of 8)
      int gr = m0 + r;
#if HAVE_ASYNC_LDS
      const bf16_t* gsrc = (gr < M) ? (A + (size_t)gr * lda + k0 + c) : zsrc;
      __builtin_amdgcn_global_load_async_to_lds_b128(
          AS1(gsrc), AS3(&As[buf][r][c]), 0, 0);
#else
      v8bfx val = zero8;
      if (gr < M) val = *(const v8bfx*)(A + (size_t)gr * lda + k0 + c);
      *(v8bfx*)(&As[buf][r][c]) = val;
#endif
    }
    // B tile: 32 rows x 64 cols = 256 x 16B, 1 per thread
    {
      int r = tid >> 3;
      int c = (tid & 7) << 3;
      const bf16_t* gsrc = B + (size_t)(k0 + r) * ldb + n0 + c;
#if HAVE_ASYNC_LDS
      __builtin_amdgcn_global_load_async_to_lds_b128(
          AS1(gsrc), AS3(&Bs[buf][r][c]), 0, 0);
#else
      *(v8bfx*)(&Bs[buf][r][c]) = *(const v8bfx*)gsrc;
#endif
    }
  };

  auto tile_sync = [&]() {
#if HAVE_ASYNC_LDS
    __builtin_amdgcn_s_wait_asynccnt(0);  // own async LDS writes done
#endif
    __syncthreads();                      // all waves' writes visible
  };

  auto compute = [&](int buf) {
    // A fragment (ISA 16x32 bf16 layout):
    //  lanes 0-15 : row=lane,    j<8 -> K=j,   j>=8 -> K=j+8
    //  lanes16-31 : row=lane-16, j<8 -> K=8+j, j>=8 -> K=16+j
    v16bfx af;
    const int arow = wave * 16 + (lane & 15);
    const int kb   = (lane < 16) ? 0 : 8;
#pragma unroll
    for (int j = 0; j < 16; ++j) {
      int kk = kb + ((j < 8) ? j : (j + 8));
      af[j] = As[buf][arow][kk];
    }
    // B fragment: lane holds column N=lane&15, elems j follow A's K order
#pragma unroll
    for (int nt = 0; nt < 4; ++nt) {
      v16bfx bfr;
#pragma unroll
      for (int j = 0; j < 16; ++j) bfr[j] = Bs[buf][kb + ((j < 8) ? j : (j + 8))][nt * 16 + (lane & 15)];
      acc[nt] = __builtin_amdgcn_wmma_f32_16x16x32_bf16(
          false, af, false, bfr, (short)0, acc[nt], false, false);
    }
  };

  // ---- peeled, double-buffered K loop ----
  stage(0, 0);
  tile_sync();
  int cur = 0;
  int k0 = 0;
  for (; k0 + BK < K; k0 += BK) {
    if (k0 + 2 * BK < K) {  // prefetch tile after next (global_prefetch_b8)
      int pr = m0 + (tid & 127);
      if (pr < M) __builtin_prefetch(A + (size_t)pr * lda + k0 + 2 * BK, 0, 1);
    }
    stage(cur ^ 1, k0 + BK);   // unconditional in-loop staging
    compute(cur);
    tile_sync();
    cur ^= 1;
  }
  compute(cur);                // last tile

  // epilogue; C/D layout: vgpr r, lanes 0-15 -> (M=r, N=lane),
  //                       lanes 16-31 -> (M=r+8, N=lane-16)
  const int lrow = (lane < 16) ? 0 : 8;
  const int lcol = lane & 15;
#pragma unroll
  for (int nt = 0; nt < 4; ++nt) {
    int gc = n0 + nt * 16 + lcol;        // always < padded N
    float bv = bias[gc];
#pragma unroll
    for (int r = 0; r < 8; ++r) {
      int gr = wm + lrow + r;
      if (gr >= M) continue;
      float v = acc[nt][r] + bv;
      if (mode == 0) {
        v = fmaxf(v, 0.0f);
        Cb[(size_t)gr * ldc + gc] = (bf16_t)v;
      } else {
        Cf[(size_t)gr * ldc + gc] = v;
      }
    }
  }
}

// ======================================================================
// padding / conversion kernels
// ======================================================================
__global__ __launch_bounds__(256)
void pad_bf16_kernel(const float* __restrict__ src, bf16_t* __restrict__ dst,
                     int rows, int cols, int rowsP, int colsP) {
  int idx = blockIdx.x * 256 + threadIdx.x;
  if (idx >= rowsP * colsP) return;
  int r = idx / colsP, c = idx % colsP;
  dst[idx] = (r < rows && c < cols) ? (bf16_t)src[r * cols + c] : (bf16_t)0.0f;
}

__global__ __launch_bounds__(256)
void pad_f32_kernel(const float* __restrict__ src, float* __restrict__ dst,
                    int rows, int cols, int rowsP, int colsP) {
  int idx = blockIdx.x * 256 + threadIdx.x;
  if (idx >= rowsP * colsP) return;
  int r = idx / colsP, c = idx % colsP;
  dst[idx] = (r < rows && c < cols) ? src[r * cols + c] : 0.0f;
}

__global__ __launch_bounds__(256)
void to_bf16_kernel(const float* __restrict__ src, bf16_t* __restrict__ dst,
                    int n) {
  int idx = blockIdx.x * 256 + threadIdx.x;
  if (idx < n) dst[idx] = (bf16_t)src[idx];
}

// ======================================================================
// graph / elementwise kernels (activation rows stride DIMP, pad cols == 0)
// ======================================================================
__global__ __launch_bounds__(256)
void embed_kernel(const int* __restrict__ x, const float* __restrict__ e1,
                  const float* __restrict__ e2, float* __restrict__ h) {
  int idx = blockIdx.x * 256 + threadIdx.x;
  if (idx >= N_NODES * DIMP) return;
  int i = idx / DIMP, d = idx % DIMP;
  h[idx] = (d < DIM)
               ? e1[x[i * 2 + 0] * DIM + d] + e2[x[i * 2 + 1] * DIM + d]
               : 0.0f;
}

__global__ __launch_bounds__(256)
void aggr_init_kernel(const float* __restrict__ h,
                      const float* __restrict__ ee1l,
                      const float* __restrict__ ee2l,
                      float* __restrict__ aggr) {
  int idx = blockIdx.x * 256 + threadIdx.x;
  if (idx >= N_NODES * DIMP) return;
  int d = idx % DIMP;
  aggr[idx] = (d < DIM) ? h[idx] + ee1l[4 * DIM + d] + ee2l[0 * DIM + d] : 0.0f;
}

__global__ __launch_bounds__(256)
void scatter_kernel(const float* __restrict__ h, const int* __restrict__ ei,
                    const int* __restrict__ ea, const float* __restrict__ ee1l,
                    const float* __restrict__ ee2l, float* __restrict__ aggr) {
  int e = blockIdx.x * 8 + (threadIdx.x >> 5);
  if (e >= N_EDGES) return;
  int lane = threadIdx.x & 31;
  int src = ei[e];
  int dst = ei[N_EDGES + e];
  int a0 = ea[e * 2 + 0], a1 = ea[e * 2 + 1];
  const float* hs = h + (size_t)src * DIMP;
  const float* p1 = ee1l + a0 * DIM;
  const float* p2 = ee2l + a1 * DIM;
  float* outp = aggr + (size_t)dst * DIMP;
  for (int d = lane; d < DIM; d += 32)
    atomicAdd(outp + d, hs[d] + p1[d] + p2[d]);
}

__global__ __launch_bounds__(256)
void bn_stats_kernel(const float* __restrict__ h, float* __restrict__ sums,
                     int n) {
  int r0 = blockIdx.x * 256;
  int rend = r0 + 256; if (rend > n) rend = n;
  int c0 = threadIdx.x;        // 0..255 < DIM
  int c1 = threadIdx.x + 256;  // valid only when < DIM
  float s0 = 0.f, q0 = 0.f, s1 = 0.f, q1 = 0.f;
  for (int r = r0; r < rend; ++r) {
    const float* row = h + (size_t)r * DIMP;
    { float v = row[c0]; s0 += v; q0 += v * v; }
    if (c1 < DIM) { float v = row[c1]; s1 += v; q1 += v * v; }
  }
  atomicAdd(&sums[c0], s0);
  atomicAdd(&sums[DIM + c0], q0);
  if (c1 < DIM) { atomicAdd(&sums[c1], s1); atomicAdd(&sums[DIM + c1], q1); }
}

__global__ __launch_bounds__(256)
void bn_norm_kernel(float* __restrict__ h, const float* __restrict__ sums,
                    const float* __restrict__ g, const float* __restrict__ b,
                    int n, int relu) {
  int idx = blockIdx.x * 256 + threadIdx.x;
  if (idx >= n * DIMP) return;
  int d = idx % DIMP;
  if (d >= DIM) return;  // pad columns stay 0
  float invn = 1.0f / (float)n;
  float mean = sums[d] * invn;
  float var = sums[DIM + d] * invn - mean * mean;
  float v = (h[idx] - mean) * rsqrtf(var + BN_EPS) * g[d] + b[d];
  if (relu) v = fmaxf(v, 0.0f);
  h[idx] = v;
}

__global__ __launch_bounds__(256)
void gather_bf16_kernel(const float* __restrict__ rep,
                        const int* __restrict__ didx,
                        bf16_t* __restrict__ A) {
  int idx = blockIdx.x * 256 + threadIdx.x;
  if (idx >= G_GRAPHS * DIMP) return;
  int g = idx / DIMP, d = idx % DIMP;
  A[idx] = (bf16_t)rep[(size_t)didx[g] * DIMP + d];  // pads are zero
}

__global__ __launch_bounds__(256)
void pool_kernel(const float* __restrict__ out, const int* __restrict__ batch,
                 float* __restrict__ feat, float* __restrict__ counts) {
  int i = blockIdx.x * 8 + (threadIdx.x >> 5);
  if (i >= N_NODES) return;
  int lane = threadIdx.x & 31;
  int g = batch[i];
  const float* row = out + (size_t)i * DIMP;
  float* frow = feat + (size_t)g * DIMP;
  for (int d = lane; d < DIM; d += 32) atomicAdd(&frow[d], row[d]);
  if (lane == 0) atomicAdd(&counts[g], 1.0f);
}

__global__ __launch_bounds__(256)
void feat_final_kernel(float* __restrict__ feat, const float* __restrict__ counts,
                       const float* __restrict__ dangout) {
  int idx = blockIdx.x * 256 + threadIdx.x;
  if (idx >= G_GRAPHS * DIMP) return;
  int g = idx / DIMP;
  feat[idx] = feat[idx] / fmaxf(counts[g], 1.0f) + dangout[idx];
}

__global__ __launch_bounds__(256)
void cls_input_kernel(const float* __restrict__ f0, const float* __restrict__ f1,
                      bf16_t* __restrict__ A) {
  int idx = blockIdx.x * 256 + threadIdx.x;
  if (idx >= 2 * G_GRAPHS * DIMP) return;
  int row = idx / DIMP, d = idx % DIMP;
  float a, b;
  if (row < G_GRAPHS) {
    a = f0[row * DIMP + d];
    b = f1[row * DIMP + d];
  } else {
    int g = row - G_GRAPHS;
    a = f0[g * DIMP + d];
    b = f1[((g - 1 + G_GRAPHS) % G_GRAPHS) * DIMP + d];
  }
  A[idx] = (bf16_t)fmaxf(a, b);
}

__global__ __launch_bounds__(256)
void cls_out_kernel(const bf16_t* __restrict__ Hbuf, const float* __restrict__ w2p,
                    const float* __restrict__ b2, float* __restrict__ out) {
  int row = blockIdx.x * 8 + (threadIdx.x >> 5);
  if (row >= 2 * G_GRAPHS) return;
  int lane = threadIdx.x & 31;
  float s = 0.0f;
  const bf16_t* hr = Hbuf + (size_t)row * DIMP;
  for (int d = lane; d < DIMP; d += 32) s += (float)hr[d] * w2p[d];  // pads 0
#pragma unroll
  for (int off = 16; off > 0; off >>= 1) s += __shfl_down(s, off, 32);
  if (lane == 0) out[row] = s + b2[0];
}

__global__ __launch_bounds__(256)
void labels_kernel(float* __restrict__ out) {
  int i = blockIdx.x * 256 + threadIdx.x;
  if (i < 2 * G_GRAPHS) out[2 * G_GRAPHS + i] = (i < G_GRAPHS) ? 1.0f : 0.0f;
}

// ======================================================================
// host launcher
// ======================================================================
static void launch_gemm(const bf16_t* A, int lda, const bf16_t* B, int ldb,
                        const float* bias, const bf16_t* zsrc,
                        int M, int K, int N, void* C,
                        int ldc, int mode, hipStream_t s) {
  dim3 grid(N / BN, ceildiv_i(M, BM));
  gemm_bf16_wmma_kernel<<<grid, 256, 0, s>>>(
      A, lda, B, ldb, bias, zsrc, M, K, N,
      mode == 0 ? (bf16_t*)C : nullptr, mode == 1 ? (float*)C : nullptr,
      ldc, mode);
}

extern "C" void kernel_launch(void* const* d_in, const int* in_sizes, int n_in,
                              void* d_out, int out_size, void* d_ws,
                              size_t ws_size, hipStream_t stream) {
  (void)in_sizes; (void)n_in; (void)out_size; (void)ws_size;

  const float* x_emb1    = (const float*)d_in[10];
  const float* x_emb2    = (const float*)d_in[11];
  const float* edge_emb1 = (const float*)d_in[12];
  const float* edge_emb2 = (const float*)d_in[13];
  const float* mlp_w1    = (const float*)d_in[14];
  const float* mlp_b1    = (const float*)d_in[15];
  const float* mlp_w2    = (const float*)d_in[16];
  const float* mlp_b2    = (const float*)d_in[17];
  const float* bn_g      = (const float*)d_in[18];
  const float* bn_b      = (const float*)d_in[19];
  const float* proj_w1   = (const float*)d_in[20];
  const float* proj_b1   = (const float*)d_in[21];
  const float* dang_w1   = (const float*)d_in[22];
  const float* dang_b1   = (const float*)d_in[23];
  const float* cls_w1    = (const float*)d_in[24];
  const float* cls_b1    = (const float*)d_in[25];
  const float* proj_w2   = (const float*)d_in[26];
  const float* proj_b2   = (const float*)d_in[27];
  const float* dang_w2   = (const float*)d_in[28];
  const float* dang_b2   = (const float*)d_in[29];
  const float* cls_w2    = (const float*)d_in[30];
  const float* cls_b2    = (const float*)d_in[31];

  char* ws = (char*)d_ws;
  size_t off = 0;
  auto carve = [&](size_t bytes) -> char* {
    char* p = ws + off;
    off += (bytes + 255) & ~(size_t)255;
    return p;
  };
  float*  h       = (float*)carve((size_t)N_NODES * DIMP * 4);
  char*   big2    = carve((size_t)N_NODES * HIDP * 2);   // aggr(f32,DIMP) / hidden(bf16)
  float*  aggr    = (float*)big2;
  bf16_t* hidden  = (bf16_t*)big2;
  bf16_t* Abuf    = (bf16_t*)carve((size_t)N_NODES * DIMP * 2);
  bf16_t* w1p     = (bf16_t*)carve((size_t)LAYERS * DIMP * HIDP * 2);
  bf16_t* w2p     = (bf16_t*)carve((size_t)LAYERS * HIDP * DIMP * 2);
  bf16_t* pw1p    = (bf16_t*)carve((size_t)DIMP * DIMP * 2);
  bf16_t* pw2p    = (bf16_t*)carve((size_t)DIMP * DIMP * 2);
  bf16_t* dw1p    = (bf16_t*)carve((size_t)DIMP * DIMP * 2);
  bf16_t* dw2p    = (bf16_t*)carve((size_t)DIMP * DIMP * 2);
  bf16_t* cw1p    = (bf16_t*)carve((size_t)DIMP * DIMP * 2);
  float*  b1p     = (float*)carve((size_t)LAYERS * HIDP * 4);
  float*  b2p     = (float*)carve((size_t)LAYERS * DIMP * 4);
  float*  pb1p    = (float*)carve(DIMP * 4);
  float*  pb2p    = (float*)carve(DIMP * 4);
  float*  db1p    = (float*)carve(DIMP * 4);
  float*  db2p    = (float*)carve(DIMP * 4);
  float*  cb1p    = (float*)carve(DIMP * 4);
  float*  cw2p    = (float*)carve(DIMP * 4);
  bf16_t* zsrc    = (bf16_t*)carve(256);                 // 16B zeros for async OOB
  float*  stats   = (float*)carve(2 * DIM * 4);
  float*  feat0   = (float*)carve((size_t)G_GRAPHS * DIMP * 4);
  float*  feat1   = (float*)carve((size_t)G_GRAPHS * DIMP * 4);
  float*  counts  = (float*)carve((size_t)G_GRAPHS * 4);
  bf16_t* dangA   = (bf16_t*)carve((size_t)G_GRAPHS * DIMP * 2);
  bf16_t* danghid = (bf16_t*)carve((size_t)G_GRAPHS * DIMP * 2);
  float*  dangout = (float*)carve((size_t)G_GRAPHS * DIMP * 4);
  bf16_t* clsA    = (bf16_t*)carve((size_t)2 * G_GRAPHS * DIMP * 2);
  bf16_t* clshid  = (bf16_t*)carve((size_t)2 * G_GRAPHS * DIMP * 2);

  const int NDP = N_NODES * DIMP;
  const int gNDP = ceildiv_i(NDP, 256);
  const int gSq = ceildiv_i(DIMP * DIMP, 256);

  (void)hipMemsetAsync(zsrc, 0, 256, stream);

  // ---- pad weights/biases into workspace (deterministic each call) ----
  for (int l = 0; l < LAYERS; ++l) {
    pad_bf16_kernel<<<ceildiv_i(DIMP * HIDP, 256), 256, 0, stream>>>(
        mlp_w1 + (size_t)l * DIM * HID, w1p + (size_t)l * DIMP * HIDP,
        DIM, HID, DIMP, HIDP);
    pad_bf16_kernel<<<ceildiv_i(HIDP * DIMP, 256), 256, 0, stream>>>(
        mlp_w2 + (size_t)l * HID * DIM, w2p + (size_t)l * HIDP * DIMP,
        HID, DIM, HIDP, DIMP);
  }
  pad_bf16_kernel<<<gSq, 256, 0, stream>>>(proj_w1, pw1p, DIM, DIM, DIMP, DIMP);
  pad_bf16_kernel<<<gSq, 256, 0, stream>>>(proj_w2, pw2p, DIM, DIM, DIMP, DIMP);
  pad_bf16_kernel<<<gSq, 256, 0, stream>>>(dang_w1, dw1p, DIM, DIM, DIMP, DIMP);
  pad_bf16_kernel<<<gSq, 256, 0, stream>>>(dang_w2, dw2p, DIM, DIM, DIMP, DIMP);
  pad_bf16_kernel<<<gSq, 256, 0, stream>>>(cls_w1, cw1p, DIM, DIM, DIMP, DIMP);
  pad_f32_kernel<<<ceildiv_i(LAYERS * HIDP, 256), 256, 0, stream>>>(
      mlp_b1, b1p, LAYERS, HID, LAYERS, HIDP);
  pad_f32_kernel<<<ceildiv_i(LAYERS * DIMP, 256), 256, 0, stream>>>(
      mlp_b2, b2p, LAYERS, DIM, LAYERS, DIMP);
  pad_f32_kernel<<<ceildiv_i(DIMP, 256), 256, 0, stream>>>(proj_b1, pb1p, 1, DIM, 1, DIMP);
  pad_f32_kernel<<<ceildiv_i(DIMP, 256), 256, 0, stream>>>(proj_b2, pb2p, 1, DIM, 1, DIMP);
  pad_f32_kernel<<<ceildiv_i(DIMP, 256), 256, 0, stream>>>(dang_b1, db1p, 1, DIM, 1, DIMP);
  pad_f32_kernel<<<ceildiv_i(DIMP, 256), 256, 0, stream>>>(dang_b2, db2p, 1, DIM, 1, DIMP);
  pad_f32_kernel<<<ceildiv_i(DIMP, 256), 256, 0, stream>>>(cls_b1, cb1p, 1, DIM, 1, DIMP);
  pad_f32_kernel<<<ceildiv_i(DIMP, 256), 256, 0, stream>>>(cls_w2, cw2p, 1, DIM, 1, DIMP);

  // ---- two branches ----
  for (int br = 0; br < 2; ++br) {
    const int* x     = (const int*)d_in[br * 5 + 0];
    const int* ei    = (const int*)d_in[br * 5 + 1];
    const int* ea    = (const int*)d_in[br * 5 + 2];
    const int* batch = (const int*)d_in[br * 5 + 3];
    const int* didx  = (const int*)d_in[br * 5 + 4];
    float* feat = (br == 0) ? feat0 : feat1;

    embed_kernel<<<gNDP, 256, 0, stream>>>(x, x_emb1, x_emb2, h);

    for (int l = 0; l < LAYERS; ++l) {
      const float* ee1l = edge_emb1 + (size_t)l * 6 * DIM;
      const float* ee2l = edge_emb2 + (size_t)l * 3 * DIM;
      aggr_init_kernel<<<gNDP, 256, 0, stream>>>(h, ee1l, ee2l, aggr);
      scatter_kernel<<<ceildiv_i(N_EDGES, 8), 256, 0, stream>>>(h, ei, ea, ee1l, ee2l, aggr);
      to_bf16_kernel<<<gNDP, 256, 0, stream>>>(aggr, Abuf, NDP);
      // hidden = relu(aggr @ w1 + b1)  [N,HIDP] bf16 (aliases aggr, consumed)
      launch_gemm(Abuf, DIMP, w1p + (size_t)l * DIMP * HIDP, HIDP,
                  b1p + (size_t)l * HIDP, zsrc, N_NODES, DIMP, HIDP,
                  hidden, HIDP, 0, stream);
      // h = hidden @ w2 + b2  [N,DIMP] f32 (pad cols come out exactly 0)
      launch_gemm(hidden, HIDP, w2p + (size_t)l * HIDP * DIMP, DIMP,
                  b2p + (size_t)l * DIMP, zsrc, N_NODES, HIDP, DIMP,
                  h, DIMP, 1, stream);
      (void)hipMemsetAsync(stats, 0, 2 * DIM * sizeof(float), stream);
      bn_stats_kernel<<<ceildiv_i(N_NODES, 256), 256, 0, stream>>>(h, stats, N_NODES);
      bn_norm_kernel<<<gNDP, 256, 0, stream>>>(h, stats, bn_g + (size_t)l * DIM,
                                               bn_b + (size_t)l * DIM, N_NODES,
                                               l < LAYERS - 1 ? 1 : 0);
    }

    gather_bf16_kernel<<<ceildiv_i(G_GRAPHS * DIMP, 256), 256, 0, stream>>>(h, didx, dangA);

    to_bf16_kernel<<<gNDP, 256, 0, stream>>>(h, Abuf, NDP);
    launch_gemm(Abuf, DIMP, pw1p, DIMP, pb1p, zsrc, N_NODES, DIMP, DIMP, hidden, DIMP, 0, stream);
    launch_gemm(hidden, DIMP, pw2p, DIMP, pb2p, zsrc, N_NODES, DIMP, DIMP, h, DIMP, 1, stream);

    (void)hipMemsetAsync(feat, 0, (size_t)G_GRAPHS * DIMP * sizeof(float), stream);
    (void)hipMemsetAsync(counts, 0, (size_t)G_GRAPHS * sizeof(float), stream);
    pool_kernel<<<ceildiv_i(N_NODES, 8), 256, 0, stream>>>(h, batch, feat, counts);

    launch_gemm(dangA, DIMP, dw1p, DIMP, db1p, zsrc, G_GRAPHS, DIMP, DIMP, danghid, DIMP, 0, stream);
    launch_gemm(danghid, DIMP, dw2p, DIMP, db2p, zsrc, G_GRAPHS, DIMP, DIMP, dangout, DIMP, 1, stream);

    feat_final_kernel<<<ceildiv_i(G_GRAPHS * DIMP, 256), 256, 0, stream>>>(feat, counts, dangout);
  }

  // ---- classifier ----
  float* out = (float*)d_out;
  cls_input_kernel<<<ceildiv_i(2 * G_GRAPHS * DIMP, 256), 256, 0, stream>>>(feat0, feat1, clsA);
  launch_gemm(clsA, DIMP, cw1p, DIMP, cb1p, zsrc, 2 * G_GRAPHS, DIMP, DIMP, clshid, DIMP, 0, stream);
  cls_out_kernel<<<ceildiv_i(2 * G_GRAPHS, 8), 256, 0, stream>>>(clshid, cw2p, cls_b2, out);
  labels_kernel<<<ceildiv_i(2 * G_GRAPHS, 256), 256, 0, stream>>>(out);
}
